// SAGE_22454089023509
// MI455X (gfx1250) — matrix operand order repaired and checked by reading, not compiled, not run on hardware
//
#include <hip/hip_runtime.h>
#include <hip/hip_bf16.h>

typedef __attribute__((ext_vector_type(16))) _Float16 v16h;
typedef __attribute__((ext_vector_type(8)))  _Float16 v8h;
typedef __attribute__((ext_vector_type(8)))  float    v8f;

#define NNODES 100000
#define NEDGES 1600000
#define CH 128
#define OUTC 40
#define OUTP 48

// ---------------- utility kernels ----------------

__global__ void k_zero(float* __restrict__ p, size_t n) {
    size_t i = (size_t)blockIdx.x * blockDim.x + threadIdx.x;
    size_t stride = (size_t)gridDim.x * blockDim.x;
    for (; i < n; i += stride) p[i] = 0.0f;
}

__global__ void k_degree(const int* __restrict__ dst, float* __restrict__ deg, int E) {
    int e = blockIdx.x * blockDim.x + threadIdx.x;
    if (e < E) atomicAdd(&deg[dst[e]], 1.0f);
}

__global__ void k_invdeg(const float* __restrict__ deg, float* __restrict__ idg, int N) {
    int i = blockIdx.x * blockDim.x + threadIdx.x;
    if (i < N) idg[i] = 1.0f / fmaxf(deg[i], 1.0f);
}

// one wave per edge; lane owns 4 channels: b128 gather + 4 f32 atomics (L2-resident)
__global__ __launch_bounds__(256)
void k_scatter(const float* __restrict__ feat, const int* __restrict__ src,
               const int* __restrict__ dst, float* __restrict__ agg, int E) {
    int w    = blockIdx.x * (blockDim.x >> 5) + (threadIdx.x >> 5);
    int lane = threadIdx.x & 31;
    if (w >= E) return;
    int s = src[w], d = dst[w];
    const float4 v = *(const float4*)(feat + (size_t)s * CH + lane * 4);
    float* ap = agg + (size_t)d * CH + lane * 4;
    atomicAdd(ap + 0, v.x); atomicAdd(ap + 1, v.y);
    atomicAdd(ap + 2, v.z); atomicAdd(ap + 3, v.w);
}

// Pack f32 weight [128 x Nout] into f16 WMMA B-fragment layout:
//   tile (kt,nt): element h of lane = W[kt*32 + 8*(lane>>4) + (h&7) + 16*(h>>3)][nt*16+(lane&15)]  (zero-padded cols)
__global__ void k_pack(const float* __restrict__ W, _Float16* __restrict__ P,
                       int Nout, int ntiles) {
    int idx = blockIdx.x * blockDim.x + threadIdx.x;
    int total = 4 * ntiles * 512;
    if (idx >= total) return;
    int h = idx & 15, lane = (idx >> 4) & 31, tile = idx >> 9;
    int nt = tile % ntiles, kt = tile / ntiles;
    int k = kt * 32 + (lane >> 4) * 8 + (h & 7) + (h >> 3) * 16, n = nt * 16 + (lane & 15);
    float val = (n < Nout) ? W[k * Nout + n] : 0.0f;
    P[idx] = (_Float16)val;
}

// ---------------- fused SAGE layer: h = x@Ws + (agg*invdeg)@Wn + b [+BN+ReLU] ----------------
// block = 16 nodes, 256 threads = 8 waves; wave wv owns cols [16*wv, 16*wv+16)

template <bool BN>
__global__ __launch_bounds__(256)
void k_sage(const float* __restrict__ xin, const float* __restrict__ agg,
            const float* __restrict__ invdeg,
            const _Float16* __restrict__ WsP, const _Float16* __restrict__ WnP,
            const float* __restrict__ bias,
            const float* __restrict__ bg, const float* __restrict__ bb,
            const float* __restrict__ bm, const float* __restrict__ bv,
            float* __restrict__ out, int Nout, int ntiles) {
    // rows padded to 136 halfs (272B = 17*16B): 16B-aligned b128 ds loads, conflict-friendly
    __shared__ __align__(16) _Float16 xs[16][136];
    __shared__ __align__(16) _Float16 as[16][136];

    const int m0  = blockIdx.x * 16;
    const int tid = threadIdx.x;

    // ---- cooperative staging: 16x128 f32 rows -> f16 LDS (agg pre-scaled by 1/deg) ----
    {
        int row = tid >> 4;
        int c0  = (tid & 15) * 8;
        const float* xp = xin + (size_t)(m0 + row) * CH + c0;
        float4 a0 = *(const float4*)(xp);
        float4 a1 = *(const float4*)(xp + 4);
        float tx[8] = {a0.x, a0.y, a0.z, a0.w, a1.x, a1.y, a1.z, a1.w};
#pragma unroll
        for (int i = 0; i < 8; ++i) xs[row][c0 + i] = (_Float16)tx[i];

        float idg = invdeg[m0 + row];
        const float* gp = agg + (size_t)(m0 + row) * CH + c0;
        float4 g0 = *(const float4*)(gp);
        float4 g1 = *(const float4*)(gp + 4);
        float tg[8] = {g0.x, g0.y, g0.z, g0.w, g1.x, g1.y, g1.z, g1.w};
#pragma unroll
        for (int i = 0; i < 8; ++i) as[row][c0 + i] = (_Float16)(tg[i] * idg);
    }
    __syncthreads();

    const int wv = tid >> 5, lane = tid & 31;
    if (wv < ntiles) {                       // wave-uniform: EXEC all-ones for WMMA
        v8f c = {};
        const int r  = lane & 15;
        const int hi = lane >> 4;            // 0: lanes 0-15, 1: lanes 16-31
#pragma unroll
        for (int kt = 0; kt < 4; ++kt) {
            // A fragment (16-bit A 16x32 layout): lanes 0-15 hold K[0..7],[16..23];
            // lanes 16-31 hold K[8..15],[24..31]
            int ks1 = kt * 32 + hi * 8;
            int ks2 = kt * 32 + 16 + hi * 8;
            v8h xlo = *(const v8h*)&xs[r][ks1];
            v8h xhi = *(const v8h*)&xs[r][ks2];
            v8h alo = *(const v8h*)&as[r][ks1];
            v8h ahi = *(const v8h*)&as[r][ks2];
            v16h ax, aa;
#pragma unroll
            for (int i = 0; i < 8; ++i) {
                ax[i] = xlo[i]; ax[i + 8] = xhi[i];
                aa[i] = alo[i]; aa[i + 8] = ahi[i];
            }
            // B fragments: pre-packed, lane = N column, halfs = 16 K values (32B load)
            const size_t boff = (((size_t)kt * ntiles + wv) * 32 + lane) * 16;
            v16h bs = *(const v16h*)(WsP + boff);
            v16h bn = *(const v16h*)(WnP + boff);
            c = __builtin_amdgcn_wmma_f32_16x16x32_f16(false, ax, false, bs,
                                                       (short)0, c, false, false);
            c = __builtin_amdgcn_wmma_f32_16x16x32_f16(false, aa, false, bn,
                                                       (short)0, c, false, false);
        }
        // ---- epilogue: channel n is lane-constant in C layout ----
        const int n = wv * 16 + r;
        if (n < Nout) {
            float scale = 1.0f, shift = bias[n];
            if constexpr (BN) {
                float rs = rsqrtf(bv[n] + 1e-5f);
                scale = bg[n] * rs;
                shift = fmaf(bias[n] - bm[n], scale, bb[n]);
            }
#pragma unroll
            for (int rr = 0; rr < 8; ++rr) {
                int m = m0 + rr + hi * 8;    // C layout: VGPR rr -> M=rr (lo) / 8+rr (hi)
                float t;
                if constexpr (BN) {
                    t = fmaxf(fmaf(c[rr], scale, shift), 0.0f);
                } else {
                    t = c[rr] + shift;
                }
                out[(size_t)m * Nout + n] = t;
            }
        }
    }
}

// ---------------- log_softmax over 40 classes: one wave per node ----------------

__global__ __launch_bounds__(256)
void k_logsoftmax(const float* __restrict__ h, float* __restrict__ out, int N) {
    int w    = blockIdx.x * (blockDim.x >> 5) + (threadIdx.x >> 5);
    int lane = threadIdx.x & 31;
    if (w >= N) return;
    const float* hp = h + (size_t)w * OUTC;
    float v0 = hp[lane];
    float v1 = (lane < 8) ? hp[32 + lane] : -3.0e38f;
    float mx = fmaxf(v0, v1);
#pragma unroll
    for (int o = 16; o > 0; o >>= 1) mx = fmaxf(mx, __shfl_xor(mx, o, 32));
    float s = __expf(v0 - mx) + ((lane < 8) ? __expf(v1 - mx) : 0.0f);
#pragma unroll
    for (int o = 16; o > 0; o >>= 1) s += __shfl_xor(s, o, 32);
    float lz = __logf(s);
    float* op = out + (size_t)w * OUTC;
    op[lane] = v0 - mx - lz;
    if (lane < 8) op[32 + lane] = v1 - mx - lz;
}

// ---------------- launch ----------------

extern "C" void kernel_launch(void* const* d_in, const int* in_sizes, int n_in,
                              void* d_out, int out_size, void* d_ws, size_t ws_size,
                              hipStream_t stream) {
    const float* x   = (const float*)d_in[0];
    const int*   ei  = (const int*)d_in[1];
    const int*   src = ei;
    const int*   dst = ei + NEDGES;
    const float* Ws1 = (const float*)d_in[2];
    const float* Wn1 = (const float*)d_in[3];
    const float* b1  = (const float*)d_in[4];
    const float* Ws2 = (const float*)d_in[5];
    const float* Wn2 = (const float*)d_in[6];
    const float* b2  = (const float*)d_in[7];
    const float* Ws3 = (const float*)d_in[8];
    const float* Wn3 = (const float*)d_in[9];
    const float* b3  = (const float*)d_in[10];
    const float* g1  = (const float*)d_in[11];
    const float* be1 = (const float*)d_in[12];
    const float* m1  = (const float*)d_in[13];
    const float* v1  = (const float*)d_in[14];
    const float* g2  = (const float*)d_in[15];
    const float* be2 = (const float*)d_in[16];
    const float* m2  = (const float*)d_in[17];
    const float* v2  = (const float*)d_in[18];

    char*  ws  = (char*)d_ws;
    size_t off = 0;
    auto alloc = [&](size_t bytes) -> char* {
        char* p = ws + off;
        off += (bytes + 255) & ~(size_t)255;
        return p;
    };
    float*    deg  = (float*)alloc((size_t)NNODES * 4);
    float*    idg  = (float*)alloc((size_t)NNODES * 4);
    float*    agg  = (float*)alloc((size_t)NNODES * CH * 4);
    float*    hA   = (float*)alloc((size_t)NNODES * CH * 4);
    float*    hB   = (float*)alloc((size_t)NNODES * CH * 4);
    _Float16* Ws1P = (_Float16*)alloc(4 * 8 * 512 * 2);
    _Float16* Wn1P = (_Float16*)alloc(4 * 8 * 512 * 2);
    _Float16* Ws2P = (_Float16*)alloc(4 * 8 * 512 * 2);
    _Float16* Wn2P = (_Float16*)alloc(4 * 8 * 512 * 2);
    _Float16* Ws3P = (_Float16*)alloc(4 * 3 * 512 * 2);
    _Float16* Wn3P = (_Float16*)alloc(4 * 3 * 512 * 2);
    float*    h3   = hA;   // hA dead after layer-2 GEMM -> reuse for layer-3 output

    const int EB = NEDGES / 8;     // wave-per-edge, 8 waves/block
    const int NB = NNODES / 16;    // 6250 row-tiles
    const int SB = (NNODES + 7) / 8;

    // degree / inverse degree (fixed over all layers)
    k_zero<<<(NNODES + 255) / 256, 256, 0, stream>>>(deg, (size_t)NNODES);
    k_degree<<<(NEDGES + 255) / 256, 256, 0, stream>>>(dst, deg, NEDGES);
    k_invdeg<<<(NNODES + 255) / 256, 256, 0, stream>>>(deg, idg, NNODES);

    // pack weights into WMMA B-fragment layout (f16)
    k_pack<<<(4 * 8 * 512 + 255) / 256, 256, 0, stream>>>(Ws1, Ws1P, CH, 8);
    k_pack<<<(4 * 8 * 512 + 255) / 256, 256, 0, stream>>>(Wn1, Wn1P, CH, 8);
    k_pack<<<(4 * 8 * 512 + 255) / 256, 256, 0, stream>>>(Ws2, Ws2P, CH, 8);
    k_pack<<<(4 * 8 * 512 + 255) / 256, 256, 0, stream>>>(Wn2, Wn2P, CH, 8);
    k_pack<<<(4 * 3 * 512 + 255) / 256, 256, 0, stream>>>(Ws3, Ws3P, OUTC, 3);
    k_pack<<<(4 * 3 * 512 + 255) / 256, 256, 0, stream>>>(Wn3, Wn3P, OUTC, 3);

    // layer 1
    k_zero<<<8192, 256, 0, stream>>>(agg, (size_t)NNODES * CH);
    k_scatter<<<EB, 256, 0, stream>>>(x, src, dst, agg, NEDGES);
    k_sage<true><<<NB, 256, 0, stream>>>(x, agg, idg, Ws1P, Wn1P, b1,
                                         g1, be1, m1, v1, hA, CH, 8);
    // layer 2
    k_zero<<<8192, 256, 0, stream>>>(agg, (size_t)NNODES * CH);
    k_scatter<<<EB, 256, 0, stream>>>(hA, src, dst, agg, NEDGES);
    k_sage<true><<<NB, 256, 0, stream>>>(hA, agg, idg, Ws2P, Wn2P, b2,
                                         g2, be2, m2, v2, hB, CH, 8);
    // layer 3 (no BN/ReLU) + log_softmax
    k_zero<<<8192, 256, 0, stream>>>(agg, (size_t)NNODES * CH);
    k_scatter<<<EB, 256, 0, stream>>>(hB, src, dst, agg, NEDGES);
    k_sage<false><<<NB, 256, 0, stream>>>(hB, agg, idg, Ws3P, Wn3P, b3,
                                          nullptr, nullptr, nullptr, nullptr,
                                          h3, OUTC, 3);
    k_logsoftmax<<<SB, 256, 0, stream>>>(h3, (float*)d_out, NNODES);
}
